// MultiHeadAttention_59399397703709
// MI455X (gfx1250) — matrix operand order repaired and checked by reading, not compiled, not run on hardware
//
#include <hip/hip_runtime.h>

typedef __attribute__((ext_vector_type(16))) _Float16 v16h;
typedef __attribute__((ext_vector_type(8)))  _Float16 v8h;
typedef __attribute__((ext_vector_type(8)))  float    v8f;
typedef __attribute__((ext_vector_type(4)))  unsigned int v4u;
typedef __attribute__((ext_vector_type(8)))  int v8i;
typedef __attribute__((ext_vector_type(4)))  int v4i;

#define WMMA_F16(a, b, c) \
  __builtin_amdgcn_wmma_f32_16x16x32_f16(false, (a), false, (b), (short)0, (c), false, false)

#define LDSW 48  // padded f16 row stride (96B: 16B-aligned, bank-conflict-breaking)

__device__ __forceinline__ v16h combine16(v8h lo, v8h hi8) {
  v16h r;
#pragma unroll
  for (int i = 0; i < 8; ++i) { r[i] = lo[i]; r[i + 8] = hi8[i]; }
  return r;
}

// A-operand fragment (16x32 f16): lanes<16 hold K 0-7/16-23, lanes>=16 K 8-15/24-31.
__device__ __forceinline__ v16h frag_a(const _Float16* rowp, int hi) {
  const _Float16* p = rowp + hi * 8;
  return combine16(*(const v8h*)p, *(const v8h*)(p + 16));
}

// B-operand fragment (32x16 f16): lanes<16 hold K 0-7/16-23, lanes>=16 K 8-15/24-31.
__device__ __forceinline__ v16h frag_b(const _Float16* rowp, int hi) {
  const _Float16* p = rowp + hi * 8;
  return combine16(*(const v8h*)p, *(const v8h*)(p + 16));
}

// Raw LDS byte offset of a __shared__ object (generic -> addrspace(3) cast).
typedef __attribute__((address_space(3))) const void* lds_cptr_t;
__device__ __forceinline__ unsigned lds_off(const void* p) {
  return (unsigned)(unsigned long long)(lds_cptr_t)p;
}

// ---------------------------------------------------------------------------
// Tensor Data Mover: async DMA of a 2D tile (32 K-elems x 128 rows, f16) into
// LDS with hardware padding: every 16 dwords (one 32-half row) insert 8 dwords
// -> LDS row stride 48 halves, matching frag_a/frag_b expectations.
// ---------------------------------------------------------------------------
__device__ __forceinline__ void tdm_load_tile_f16(
    unsigned lds_byte, const _Float16* gptr, int dim0 /*tensor K*/,
    int dim1 /*tensor rows*/, long long stride0 /*elements*/)
{
  const unsigned long long ga = (unsigned long long)gptr;
  const v4u g0 = {
      1u,                                       // count=1, user mode
      lds_byte,                                 // lds_addr
      (unsigned)ga,                             // global_addr[31:0]
      (unsigned)((ga >> 32) & 0x1ffffffu) | (2u << 30)  // addr[56:32] | type=2
  };
  const unsigned s_lo = (unsigned)(stride0 & 0xffffffffll);
  const unsigned s_hi = (unsigned)((stride0 >> 32) & 0xffffll);
  const v8i g1 = {
      (int)((1u << 16) |        // data_size = 2 bytes
            (1u << 20) |        // pad_enable
            (3u << 22) |        // pad_interval = 16 dwords
            (7u << 25)),        // pad_amount   = 8 dwords
      (int)(((unsigned)dim0 & 0xffffu) << 16),                    // tensor_dim0[15:0]
      (int)((((unsigned)dim0 >> 16) & 0xffffu) |
            (((unsigned)dim1 & 0xffffu) << 16)),                  // dim0[31:16] | dim1[15:0]
      (int)((((unsigned)dim1 >> 16) & 0xffffu) | (32u << 16)),    // dim1[31:16] | tile_dim0=32
      (int)(128u),                                                // tile_dim1=128, tile_dim2=0
      (int)s_lo,                                                  // tensor_dim0_stride[31:0]
      (int)s_hi,                                                  // stride0[47:32] | dim1_stride lo = 0
      0                                                           // dim1_stride hi
  };
  const v4i z4 = {0, 0, 0, 0};
  const v8i z8 = {0, 0, 0, 0, 0, 0, 0, 0};
  __builtin_amdgcn_tensor_load_to_lds(g0, g1, z4, z4, z8, 0);
}

// ---------------------------------------------------------------------------
// C = A(f16, MxK row-major) @ Bt^T (Bt = f16 NxK row-major), f32 accumulate.
// Output: f16 (Ch) or f32 + bias (C).  Double-buffered TDM tile pipeline.
// Block tile 128x128, BK=32, 256 threads (8 waves: 4 along M, 2 along N).
// ---------------------------------------------------------------------------
__global__ __launch_bounds__(256) void gemm_f16_tdm(
    const _Float16* __restrict__ A, const _Float16* __restrict__ Bt,
    float* __restrict__ C, _Float16* __restrict__ Ch,
    const float* __restrict__ bias, int M, int N, int K)
{
  __shared__ __align__(16) _Float16 As[2][128][LDSW];  // [buf][m][k]
  __shared__ __align__(16) _Float16 Bs[2][128][LDSW];  // [buf][n][k]

  const int tid  = threadIdx.x;
  const int wave = tid >> 5, lane = tid & 31;
  const int l15  = lane & 15, hi = lane >> 4;
  const int wm   = wave & 3;   // 0..3 -> 32-row band
  const int wn   = wave >> 2;  // 0..1 -> 64-col band
  const int bm   = blockIdx.y * 128;
  const int bn   = blockIdx.x * 128;

  const unsigned asBase = lds_off(&As[0][0][0]);
  const unsigned bsBase = lds_off(&Bs[0][0][0]);
  const unsigned bufBytes = 128 * LDSW * 2;

  v8f acc[2][4] = {};

  const int nkb = K >> 5;
  if (wave == 0) {
    tdm_load_tile_f16(asBase, A  + (size_t)bm * K, K, M, K);
    tdm_load_tile_f16(bsBase, Bt + (size_t)bn * K, K, N, K);
  }

  for (int kb = 0; kb < nkb; ++kb) {
    const int buf = kb & 1;
    if (wave == 0) __builtin_amdgcn_s_wait_tensorcnt(0);
    __syncthreads();  // tiles for this iteration are resident

    if (wave == 0 && kb + 1 < nkb) {  // kick off next pair while computing
      const int k1 = (kb + 1) << 5;
      const unsigned nb = (unsigned)(buf ^ 1) * bufBytes;
      tdm_load_tile_f16(asBase + nb, A  + (size_t)bm * K + k1, K, M, K);
      tdm_load_tile_f16(bsBase + nb, Bt + (size_t)bn * K + k1, K, N, K);
    }

    v16h af[2], bf[4];
#pragma unroll
    for (int mt = 0; mt < 2; ++mt)
      af[mt] = frag_a(&As[buf][wm * 32 + mt * 16 + l15][0], hi);
#pragma unroll
    for (int nt = 0; nt < 4; ++nt)
      bf[nt] = frag_b(&Bs[buf][wn * 64 + nt * 16 + l15][0], hi);
#pragma unroll
    for (int mt = 0; mt < 2; ++mt)
#pragma unroll
      for (int nt = 0; nt < 4; ++nt)
        acc[mt][nt] = WMMA_F16(af[mt], bf[nt], acc[mt][nt]);

    __syncthreads();  // all reads done before this buffer is DMA-overwritten
  }

#pragma unroll
  for (int mt = 0; mt < 2; ++mt)
#pragma unroll
    for (int nt = 0; nt < 4; ++nt) {
      const int gcol = bn + wn * 64 + nt * 16 + l15;
      const float bv = bias ? bias[gcol] : 0.0f;
#pragma unroll
      for (int r = 0; r < 8; ++r) {
        const int grow = bm + wm * 32 + mt * 16 + r + 8 * hi;
        if (Ch) Ch[(size_t)grow * N + gcol] = (_Float16)acc[mt][nt][r];
        else    C [(size_t)grow * N + gcol] = acc[mt][nt][r] + bv;
      }
    }
}

// ---------------------------------------------------------------------------
// Elementwise f32 -> f16.
// ---------------------------------------------------------------------------
__global__ __launch_bounds__(256) void conv_to_f16(
    const float* __restrict__ in, _Float16* __restrict__ out, int total)
{
  const int e = blockIdx.x * 256 + threadIdx.x;
  if (e < total) out[e] = (_Float16)in[e];
}

// ---------------------------------------------------------------------------
// Tiled transpose + convert: out(CxR, f16) = in(RxC, f32)^T.
// ---------------------------------------------------------------------------
__global__ __launch_bounds__(256) void transpose_to_f16(
    const float* __restrict__ in, _Float16* __restrict__ out, int R, int Cc)
{
  __shared__ float t[32][33];
  const int bx = blockIdx.x * 32;  // col offset in `in`
  const int by = blockIdx.y * 32;  // row offset in `in`
  const int tx = threadIdx.x & 31, ty = threadIdx.x >> 5;  // 32x8
#pragma unroll
  for (int i = 0; i < 4; ++i)
    t[ty + 8 * i][tx] = in[(size_t)(by + ty + 8 * i) * Cc + bx + tx];
  __syncthreads();
#pragma unroll
  for (int i = 0; i < 4; ++i)
    out[(size_t)(bx + ty + 8 * i) * R + by + tx] = (_Float16)t[tx][ty + 8 * i];
}

// ---------------------------------------------------------------------------
// RoPE, f16 -> f16, pos = token % S, layout (token, rowlen).
// ---------------------------------------------------------------------------
__global__ __launch_bounds__(256) void rope_f16(
    const _Float16* __restrict__ in, _Float16* __restrict__ out,
    int total, int S, int rowlen)
{
  const int e = blockIdx.x * 256 + threadIdx.x;
  if (e >= total) return;
  const int token = e / rowlen;
  const int s = token % S;
  const int d = e & 127;
  const int i = d & 63;
  const float theta = (float)s * __expf(-(float)i * (9.210340371976184f / 64.0f));
  float cs, sn;
  __sincosf(theta, &sn, &cs);
  const float x = (float)in[e];
  const float other = (d < 64) ? -(float)in[e + 64] : (float)in[e - 64];
  out[e] = (_Float16)(x * cs + other * sn);
}

// ---------------------------------------------------------------------------
// Causal GQA flash attention. Q (B,S,H,128), K/V (B,S,KVH,128) f16; out f16.
// Grid (S/128, H, B); 256 threads = 8 waves; wave owns 16 q-rows.
// ---------------------------------------------------------------------------
__global__ __launch_bounds__(256) void flash_attn_wmma(
    const _Float16* __restrict__ Qh, const _Float16* __restrict__ Kh,
    const _Float16* __restrict__ Vh, _Float16* __restrict__ Oh, int S)
{
  constexpr int H = 32, KVH = 8, DH = 128;
  constexpr float SCALE = 0.08838834764831845f;  // 1/sqrt(128)

  __shared__ __align__(16) _Float16 Vt[128][LDSW];    // [d][key] transposed V tile
  __shared__ __align__(16) _Float16 Pt[8][16][LDSW];  // per-wave P (row, key)

  const int tid  = threadIdx.x;
  const int wave = tid >> 5, lane = tid & 31;
  const int l15  = lane & 15, hi = lane >> 4;
  const int qblk = blockIdx.x, h = blockIdx.y, b = blockIdx.z;
  const int kvh  = h >> 2;  // N_REP = 4
  const int qrow0 = qblk * 128 + wave * 16;
  const int qmax  = qrow0 + 15;

  const _Float16* qbase = Qh + (((size_t)b * S + qrow0 + l15) * H + h) * DH;
  v16h qf[4];
#pragma unroll
  for (int ks = 0; ks < 4; ++ks) qf[ks] = frag_a(qbase + ks * 32, hi);

  v8f oacc[8] = {};
  float m[8], l[8];
#pragma unroll
  for (int r = 0; r < 8; ++r) { m[r] = -1e30f; l[r] = 0.0f; }

  const int nkb = (qblk + 1) * 4;  // 32-key blocks covering [0, (qblk+1)*128)
  for (int kb = 0; kb < nkb; ++kb) {
    const int k0 = kb * 32;

    // cooperative V tile load -> transposed LDS [d][key]
#pragma unroll
    for (int i = 0; i < 2; ++i) {
      const int c = tid + 256 * i;  // 512 v8h chunks
      const int key = c >> 4;
      const int dch = (c & 15) << 3;
      const v8h vv = *(const v8h*)(Vh + (((size_t)b * S + k0 + key) * KVH + kvh) * DH + dch);
#pragma unroll
      for (int j = 0; j < 8; ++j) Vt[dch + j][key] = vv[j];
    }
    __syncthreads();

    if (k0 <= qmax) {  // causal: skip fully-masked key tiles
      v8f sacc[2] = {};
#pragma unroll
      for (int t = 0; t < 2; ++t) {
        const _Float16* kbase =
            Kh + (((size_t)b * S + k0 + t * 16 + l15) * KVH + kvh) * DH;
#pragma unroll
        for (int ks = 0; ks < 4; ++ks) {
          const v16h bf = frag_b(kbase + ks * 32, hi);
          sacc[t] = WMMA_F16(qf[ks], bf, sacc[t]);
        }
      }

      // online softmax: mask, row-max/row-sum via 16-lane butterflies
#pragma unroll
      for (int r = 0; r < 8; ++r) {
        const int row = qrow0 + r + 8 * hi;
        float s0 = sacc[0][r] * SCALE;
        float s1 = sacc[1][r] * SCALE;
        if (k0 + l15 > row)      s0 = -1e30f;
        if (k0 + 16 + l15 > row) s1 = -1e30f;
        float mx = fmaxf(s0, s1);
#pragma unroll
        for (int msk = 1; msk < 16; msk <<= 1) mx = fmaxf(mx, __shfl_xor(mx, msk, 32));
        const float mnew = fmaxf(m[r], mx);
        const float corr = __expf(m[r] - mnew);
        const float p0 = __expf(s0 - mnew);
        const float p1 = __expf(s1 - mnew);
        float rs = p0 + p1;
#pragma unroll
        for (int msk = 1; msk < 16; msk <<= 1) rs += __shfl_xor(rs, msk, 32);
        l[r] = l[r] * corr + rs;
        m[r] = mnew;
#pragma unroll
        for (int nt = 0; nt < 8; ++nt) oacc[nt][r] *= corr;
        Pt[wave][r + 8 * hi][l15]      = (_Float16)p0;
        Pt[wave][r + 8 * hi][16 + l15] = (_Float16)p1;
      }

      // O += P @ V
      const v16h pf = frag_a(&Pt[wave][l15][0], hi);
#pragma unroll
      for (int nt = 0; nt < 8; ++nt) {
        const v16h vf = frag_b(&Vt[nt * 16 + l15][0], hi);
        oacc[nt] = WMMA_F16(pf, vf, oacc[nt]);
      }
    }
    __syncthreads();
  }

  float inv[8];
#pragma unroll
  for (int r = 0; r < 8; ++r) inv[r] = 1.0f / l[r];
  _Float16* obase = Oh + ((size_t)b * S + qrow0) * (H * DH) + h * DH;
#pragma unroll
  for (int nt = 0; nt < 8; ++nt)
#pragma unroll
    for (int r = 0; r < 8; ++r) {
      const int rr = r + 8 * hi;
      obase[(size_t)rr * (H * DH) + nt * 16 + l15] = (_Float16)(oacc[nt][r] * inv[r]);
    }
}

// ---------------------------------------------------------------------------
extern "C" void kernel_launch(void* const* d_in, const int* in_sizes, int n_in,
                              void* d_out, int out_size, void* d_ws, size_t ws_size,
                              hipStream_t stream) {
  (void)in_sizes; (void)n_in; (void)out_size; (void)ws_size;
  const float* x  = (const float*)d_in[0];   // (2,2048,4096)
  const float* Wq = (const float*)d_in[1];   // (4096,4096)
  const float* Wk = (const float*)d_in[2];   // (4096,1024)
  const float* Wv = (const float*)d_in[3];   // (4096,1024)
  const float* Wo = (const float*)d_in[4];   // (4096,4096)
  const float* bo = (const float*)d_in[5];   // (4096,)
  float* out = (float*)d_out;

  constexpr int S = 2048, D = 4096, Btok = 4096;  // B*S tokens
  constexpr int Nq = 4096, Nkv = 1024;
  constexpr size_t QE = (size_t)Btok * Nq;   // 16M elements
  constexpr size_t KE = (size_t)Btok * Nkv;  // 4M elements

  _Float16* xh   = (_Float16*)d_ws;
  _Float16* Wqt  = xh + QE;          // 4096 x 4096 (N x K)
  _Float16* Wkt  = Wqt + QE;         // 1024 x 4096
  _Float16* Wvt  = Wkt + KE;         // 1024 x 4096
  _Float16* Wot  = Wvt + KE;         // 4096 x 4096
  _Float16* Qraw = Wot + QE;
  _Float16* Kraw = Qraw + QE;
  _Float16* Qh   = Kraw + KE;
  _Float16* Kh   = Qh + QE;
  _Float16* Vh   = Kh + KE;
  _Float16* Ohh  = Vh + KE;          // attention context, f16

  const dim3 blk(256);
  // f32 -> f16 staging
  conv_to_f16<<<(QE + 255) / 256, blk, 0, stream>>>(x, xh, (int)QE);
  transpose_to_f16<<<dim3(Nq / 32,  D / 32), blk, 0, stream>>>(Wq, Wqt, D, Nq);
  transpose_to_f16<<<dim3(Nkv / 32, D / 32), blk, 0, stream>>>(Wk, Wkt, D, Nkv);
  transpose_to_f16<<<dim3(Nkv / 32, D / 32), blk, 0, stream>>>(Wv, Wvt, D, Nkv);
  transpose_to_f16<<<dim3(Nq / 32,  D / 32), blk, 0, stream>>>(Wo, Wot, D, Nq);
  // QKV projections (TDM-fed WMMA GEMMs, f16 out)
  gemm_f16_tdm<<<dim3(Nq / 128,  Btok / 128), blk, 0, stream>>>(xh, Wqt, nullptr, Qraw, nullptr, Btok, Nq, D);
  gemm_f16_tdm<<<dim3(Nkv / 128, Btok / 128), blk, 0, stream>>>(xh, Wkt, nullptr, Kraw, nullptr, Btok, Nkv, D);
  gemm_f16_tdm<<<dim3(Nkv / 128, Btok / 128), blk, 0, stream>>>(xh, Wvt, nullptr, Vh,   nullptr, Btok, Nkv, D);
  // RoPE
  rope_f16<<<(QE + 255) / 256, blk, 0, stream>>>(Qraw, Qh, (int)QE, S, Nq);
  rope_f16<<<(KE + 255) / 256, blk, 0, stream>>>(Kraw, Kh, (int)KE, S, Nkv);
  // causal GQA flash attention
  flash_attn_wmma<<<dim3(S / 128, 32, 2), blk, 0, stream>>>(Qh, Kh, Vh, Ohh, S);
  // output projection + bias (fp32 out)
  gemm_f16_tdm<<<dim3(D / 128, Btok / 128), blk, 0, stream>>>(Ohh, Wot, out, nullptr, bo, Btok, D, D);
}
